// BertSelfAttention_39376260170157
// MI455X (gfx1250) — 8-bit matrix operand order repaired and checked by reading, compiled, not run on hardware
//
#include <hip/hip_runtime.h>
#include <math.h>

#define S_LEN 1024
#define H_DIM 1024
#define NHEAD 16
#define HDIM  64
#define BATCH 4
#define EPSQ  1e-5f

typedef int v8i __attribute__((ext_vector_type(8)));

__device__ __forceinline__ signed char sgn8(float x) {
    return (signed char)((x > 0.f) - (x < 0.f));
}

__global__ void k_zero(float* sums) {
    if (threadIdx.x < 3) sums[threadIdx.x] = 0.f;
}

__global__ void k_abssum(const float* __restrict__ W0, const float* __restrict__ W1,
                         const float* __restrict__ W2, float* __restrict__ sums) {
    int mat = blockIdx.x >> 8;
    int blk = blockIdx.x & 255;
    const float* W = (mat == 0) ? W0 : ((mat == 1) ? W1 : W2);
    int base = blk * 4096 + threadIdx.x;
    float s = 0.f;
    #pragma unroll
    for (int i = 0; i < 16; ++i) s += fabsf(W[base + i * 256]);
    __shared__ float red[256];
    red[threadIdx.x] = s;
    __syncthreads();
    for (int off = 128; off > 0; off >>= 1) {
        if (threadIdx.x < off) red[threadIdx.x] += red[threadIdx.x + off];
        __syncthreads();
    }
    if (threadIdx.x == 0) atomicAdd(&sums[mat], red[0]);
}

__global__ void k_pack_x(const float* __restrict__ x, signed char* __restrict__ xs, int n) {
    int i = blockIdx.x * 256 + threadIdx.x;
    if (i < n) xs[i] = sgn8(x[i]);
}

__global__ void k_pack_w(const float* __restrict__ W0, const float* __restrict__ W1,
                         const float* __restrict__ W2, signed char* __restrict__ wsg) {
    int i = blockIdx.x * 256 + threadIdx.x;
    int mat = i >> 20;
    int off = i & ((1 << 20) - 1);
    const float* W = (mat == 0) ? W0 : ((mat == 1) ? W1 : W2);
    wsg[i] = sgn8(W[off]);
}

__global__ void __launch_bounds__(256) k_qkv(
    const signed char* __restrict__ xs,
    const signed char* __restrict__ ws,
    const float* __restrict__ bq, const float* __restrict__ bk, const float* __restrict__ bv,
    const float* __restrict__ aq, const float* __restrict__ ak, const float* __restrict__ av,
    const float* __restrict__ sums,
    signed char* __restrict__ qs, signed char* __restrict__ ks_, signed char* __restrict__ vts)
{
    const int lane = threadIdx.x & 31;
    const int half = lane >> 4;
    const int nloc = lane & 15;
    const int wave = threadIdx.x >> 5;

    int gw  = blockIdx.x * 8 + wave;
    int mat = gw >> 11;
    int rem = gw & 2047;
    int mb  = rem >> 5;
    int nb  = rem & 31;

    const signed char* W    = ws + (size_t)mat * (H_DIM * (size_t)H_DIM);
    const float*       bias = (mat == 0) ? bq : ((mat == 1) ? bk : bv);
    const float*       alph = (mat == 0) ? aq : ((mat == 1) ? ak : av);
    float scale = fmaxf(alph[0], EPSQ) * (sums[mat] * (1.0f / (1024.f * 1024.f)));

    const signed char* a0 = xs + (size_t)(mb * 64 + nloc) * H_DIM + half * 16;
    const signed char* b0 = W  + (size_t)(nb * 32 + nloc) * H_DIM + half * 16;

    v8i acc[4][2] = {};
    for (int kt = 0; kt < 16; ++kt) {
        const int kb = kt * 64;
        v8i Areg[4], Breg[2];
        #pragma unroll
        for (int mi = 0; mi < 4; ++mi) {
            const signed char* ar = a0 + (size_t)(mi * 16) * H_DIM + kb;
            int4 x0 = *(const int4*)(ar);
            int4 x1 = *(const int4*)(ar + 32);
            Areg[mi][0] = x0.x; Areg[mi][1] = x0.y;
            Areg[mi][2] = x0.z; Areg[mi][3] = x0.w;
            Areg[mi][4] = x1.x; Areg[mi][5] = x1.y;
            Areg[mi][6] = x1.z; Areg[mi][7] = x1.w;
        }
        #pragma unroll
        for (int ni = 0; ni < 2; ++ni) {
            const signed char* br = b0 + (size_t)(ni * 16) * H_DIM + kb;
            int4 y0 = *(const int4*)(br);
            int4 y1 = *(const int4*)(br + 32);
            Breg[ni][0] = y0.x; Breg[ni][1] = y0.y;
            Breg[ni][2] = y0.z; Breg[ni][3] = y0.w;
            Breg[ni][4] = y1.x; Breg[ni][5] = y1.y;
            Breg[ni][6] = y1.z; Breg[ni][7] = y1.w;
        }
        #pragma unroll
        for (int ni = 0; ni < 2; ++ni)
            #pragma unroll
            for (int mi = 0; mi < 4; ++mi)
                acc[mi][ni] = __builtin_amdgcn_wmma_i32_16x16x64_iu8(
                    true, Areg[mi], true, Breg[ni], acc[mi][ni], false, false);
    }

    #pragma unroll
    for (int ni = 0; ni < 2; ++ni) {
        const int n_g = nb * 32 + ni * 16 + nloc;
        const int h = n_g >> 6, d = n_g & 63;
        const float bi = bias[n_g];
        #pragma unroll
        for (int mi = 0; mi < 4; ++mi) {
            #pragma unroll
            for (int j = 0; j < 8; ++j) {
                int m_g = mb * 64 + mi * 16 + j + half * 8;
                int b = m_g >> 10, s = m_g & 1023;
                float f = (float)acc[mi][ni][j] * scale + bi;
                signed char sg = sgn8(f);
                int bh = b * NHEAD + h;
                if (mat == 0)      qs [((size_t)(bh * S_LEN + s)) * HDIM + d] = sg;
                else if (mat == 1) ks_[((size_t)(bh * S_LEN + s)) * HDIM + d] = sg;
                else               vts[((size_t)(bh * HDIM + d)) * S_LEN + s] = sg;
            }
        }
    }
}

__global__ void __launch_bounds__(32) k_attn(
    const signed char* __restrict__ qs,
    const signed char* __restrict__ ks_,
    const signed char* __restrict__ vts,
    const float* __restrict__ mask,
    const float* __restrict__ cq, const float* __restrict__ ck,
    const float* __restrict__ cv, const float* __restrict__ ca,
    float* __restrict__ out)
{
    extern __shared__ char smem[];
    float*         sc = (float*)smem;
    unsigned char* pq = (unsigned char*)(smem + 16 * 1024 * 4);

    const int lane = threadIdx.x & 31;
    const int half = lane >> 4;
    const int nloc = lane & 15;

    int t  = blockIdx.x;
    int bh = t >> 6;
    int qt = t & 63;
    int b  = bh >> 4;
    int h  = bh & 15;

    float sq = fmaxf(cq[0], EPSQ);
    float sk = fmaxf(ck[0], EPSQ);
    float sv = fmaxf(cv[0], EPSQ);
    float sa = fmaxf(ca[0], EPSQ);
    float sscale = sq * sk * 0.125f;

    const signed char* arow = qs + ((size_t)(bh * S_LEN + qt * 16 + nloc)) * HDIM + half * 16;
    v8i A;
    {
        int4 a03 = *(const int4*)(arow);
        int4 a47 = *(const int4*)(arow + 32);
        A[0] = a03.x; A[1] = a03.y; A[2] = a03.z; A[3] = a03.w;
        A[4] = a47.x; A[5] = a47.y; A[6] = a47.z; A[7] = a47.w;
    }

    const float* mrow = mask + (size_t)b * S_LEN;
    for (int nt = 0; nt < 64; ++nt) {
        const signed char* brow = ks_ + ((size_t)(bh * S_LEN + nt * 16 + nloc)) * HDIM + half * 16;
        int4 b03 = *(const int4*)(brow);
        int4 b47 = *(const int4*)(brow + 32);
        v8i Bm;
        Bm[0] = b03.x; Bm[1] = b03.y; Bm[2] = b03.z; Bm[3] = b03.w;
        Bm[4] = b47.x; Bm[5] = b47.y; Bm[6] = b47.z; Bm[7] = b47.w;
        v8i c = {};
        c = __builtin_amdgcn_wmma_i32_16x16x64_iu8(true, A, true, Bm, c, false, false);
        int col = nt * 16 + nloc;
        #pragma unroll
        for (int j = 0; j < 8; ++j) {
            int m = j + half * 8;
            sc[m * 1024 + col] = (float)c[j] * sscale + mrow[col];
        }
    }
    __syncthreads();

    for (int r = 0; r < 16; ++r) {
        float mx = -3.4e38f;
        for (int j = 0; j < 32; ++j)
            mx = fmaxf(mx, sc[r * 1024 + lane + j * 32]);
        for (int o = 16; o > 0; o >>= 1) mx = fmaxf(mx, __shfl_xor(mx, o, 32));
        float sm = 0.f;
        for (int j = 0; j < 32; ++j) {
            int idx = r * 1024 + lane + j * 32;
            float e = __expf(sc[idx] - mx);
            sc[idx] = e;
            sm += e;
        }
        for (int o = 16; o > 0; o >>= 1) sm += __shfl_xor(sm, o, 32);
        float thr = 0.5f * sa * sm;
        for (int j = 0; j < 32; ++j) {
            int idx = r * 1024 + lane + j * 32;
            pq[idx] = (sc[idx] >= thr) ? 1 : 0;
        }
    }
    __syncthreads();

    v8i acc0 = {}, acc1 = {}, acc2 = {}, acc3 = {};
    const unsigned char* prow = pq + (size_t)nloc * 1024 + half * 16;
    for (int kt = 0; kt < 16; ++kt) {
        int kb = kt * 64;
        int4 a03 = *(const int4*)(prow + kb);
        int4 a47 = *(const int4*)(prow + kb + 32);
        v8i Ap;
        Ap[0] = a03.x; Ap[1] = a03.y; Ap[2] = a03.z; Ap[3] = a03.w;
        Ap[4] = a47.x; Ap[5] = a47.y; Ap[6] = a47.z; Ap[7] = a47.w;
        #pragma unroll
        for (int dt = 0; dt < 4; ++dt) {
            const signed char* brow =
                vts + ((size_t)(bh * HDIM + dt * 16 + nloc)) * S_LEN + kb + half * 16;
            int4 b03 = *(const int4*)(brow);
            int4 b47 = *(const int4*)(brow + 32);
            v8i Bm;
            Bm[0] = b03.x; Bm[1] = b03.y; Bm[2] = b03.z; Bm[3] = b03.w;
            Bm[4] = b47.x; Bm[5] = b47.y; Bm[6] = b47.z; Bm[7] = b47.w;
            v8i& acc = (dt == 0) ? acc0 : ((dt == 1) ? acc1 : ((dt == 2) ? acc2 : acc3));
            acc = __builtin_amdgcn_wmma_i32_16x16x64_iu8(false, Ap, true, Bm, acc, false, false);
        }
    }

    float oscale = sa * sv;
    #pragma unroll
    for (int dt = 0; dt < 4; ++dt) {
        const v8i& acc = (dt == 0) ? acc0 : ((dt == 1) ? acc1 : ((dt == 2) ? acc2 : acc3));
        int dcol = h * 64 + dt * 16 + nloc;
        #pragma unroll
        for (int j = 0; j < 8; ++j) {
            int m = j + half * 8;
            int srow = qt * 16 + m;
            out[((size_t)(b * S_LEN + srow)) * H_DIM + dcol] = (float)acc[j] * oscale;
        }
    }
}

extern "C" void kernel_launch(void* const* d_in, const int* in_sizes, int n_in,
                              void* d_out, int out_size, void* d_ws, size_t ws_size,
                              hipStream_t stream)
{
    const float* hid  = (const float*)d_in[0];
    const float* mask = (const float*)d_in[1];
    const float* Wq   = (const float*)d_in[2];
    const float* bq   = (const float*)d_in[3];
    const float* Wk   = (const float*)d_in[4];
    const float* bk   = (const float*)d_in[5];
    const float* Wv   = (const float*)d_in[6];
    const float* bv   = (const float*)d_in[7];
    const float* aq   = (const float*)d_in[8];
    const float* ak   = (const float*)d_in[9];
    const float* av   = (const float*)d_in[10];
    const float* cq   = (const float*)d_in[11];
    const float* ck   = (const float*)d_in[12];
    const float* cv   = (const float*)d_in[13];
    const float* ca   = (const float*)d_in[14];
    float* out = (float*)d_out;

    char* ws = (char*)d_ws;
    signed char* xs  = (signed char*)(ws);
    signed char* wsg = (signed char*)(ws + (size_t)(4u << 20));
    signed char* qs  = (signed char*)(ws + (size_t)(7u << 20));
    signed char* ksn = (signed char*)(ws + (size_t)(11u << 20));
    signed char* vts = (signed char*)(ws + (size_t)(15u << 20));
    float* sums      = (float*)(ws + (size_t)(19u << 20));

    k_zero  <<<1, 32, 0, stream>>>(sums);
    k_abssum<<<768, 256, 0, stream>>>(Wq, Wk, Wv, sums);
    k_pack_x<<<(BATCH * S_LEN * H_DIM) / 256, 256, 0, stream>>>(hid, xs, BATCH * S_LEN * H_DIM);
    k_pack_w<<<(3 * H_DIM * H_DIM) / 256, 256, 0, stream>>>(Wq, Wk, Wv, wsg);
    k_qkv   <<<768, 256, 0, stream>>>(xs, wsg, bq, bk, bv, aq, ak, av, sums, qs, ksn, vts);
    k_attn  <<<4096, 32, 81920, stream>>>(qs, ksn, vts, mask, cq, ck, cv, ca, out);
}
